// BitLinear_26027501814033
// MI455X (gfx1250) — matrix operand order repaired and checked by reading, not compiled, not run on hardware
//
#include <hip/hip_runtime.h>

#define IN_F   4096
#define OUT_F  4096
#define M_ROWS 8192
#define WELEMS (IN_F * OUT_F)     // 16777216
#define XELEMS (M_ROWS * IN_F)    // 33554432

#define BM 128
#define BN 256
#define BK 32
#define LDSS 40   // halves per LDS row (32 + 8 pad -> 80B stride, conflict-free b128)

typedef __attribute__((ext_vector_type(16))) _Float16 v16h;
typedef __attribute__((ext_vector_type(4)))  _Float16 v4h;
typedef __attribute__((ext_vector_type(8)))  float    v8f;
typedef int v4i __attribute__((vector_size(16)));

// ---------- CDNA5 async global->LDS copy (ASYNCcnt path), with fallback ----------
#if __has_builtin(__builtin_amdgcn_global_load_async_to_lds_b128)
#define HAVE_ASYNC 1
typedef __attribute__((address_space(3))) v4i lds_v4i;
static __device__ __forceinline__ void async_b128(const _Float16* g, _Float16* l) {
    __builtin_amdgcn_global_load_async_to_lds_b128((v4i*)g, (lds_v4i*)l, 0, 0);
}
#else
#define HAVE_ASYNC 0
#endif

static __device__ __forceinline__ void wait_async0() {
#if __has_builtin(__builtin_amdgcn_s_wait_asynccnt)
    __builtin_amdgcn_s_wait_asynccnt(0);
#else
    asm volatile("s_wait_asynccnt 0x0" ::: "memory");
#endif
}

// ---------------- gamma = max(mean(|w|), 1e-8), deterministic 2-stage ----------------

__global__ void absmean_partial(const float4* __restrict__ w, float* __restrict__ partials) {
    __shared__ float red[256];
    float s = 0.0f;
    const int n4 = WELEMS / 4;
    for (int i = blockIdx.x * 256 + threadIdx.x; i < n4; i += 256 * 256) {
        float4 v = w[i];
        s += fabsf(v.x) + fabsf(v.y) + fabsf(v.z) + fabsf(v.w);
    }
    red[threadIdx.x] = s;
    __syncthreads();
    for (int o = 128; o > 0; o >>= 1) {
        if (threadIdx.x < o) red[threadIdx.x] += red[threadIdx.x + o];
        __syncthreads();
    }
    if (threadIdx.x == 0) partials[blockIdx.x] = red[0];
}

__global__ void gamma_finalize(const float* __restrict__ partials, float* __restrict__ gamma) {
    __shared__ float red[256];
    red[threadIdx.x] = partials[threadIdx.x];
    __syncthreads();
    for (int o = 128; o > 0; o >>= 1) {
        if (threadIdx.x < o) red[threadIdx.x] += red[threadIdx.x + o];
        __syncthreads();
    }
    if (threadIdx.x == 0) gamma[0] = fmaxf(red[0] / (float)WELEMS, 1e-8f);
}

// ---------------- quantize weight -> ternary f16; cast x -> f16 ----------------

__global__ void quant_w(const float4* __restrict__ w, const float* __restrict__ gamma,
                        v4h* __restrict__ qw) {
    const int i = blockIdx.x * 256 + threadIdx.x;
    const float inv = 1.0f / gamma[0];
    float4 v = w[i];
    v4h q;
    q[0] = (_Float16)fminf(fmaxf(rintf(v.x * inv), -1.0f), 1.0f);
    q[1] = (_Float16)fminf(fmaxf(rintf(v.y * inv), -1.0f), 1.0f);
    q[2] = (_Float16)fminf(fmaxf(rintf(v.z * inv), -1.0f), 1.0f);
    q[3] = (_Float16)fminf(fmaxf(rintf(v.w * inv), -1.0f), 1.0f);
    qw[i] = q;
}

__global__ void cast_x(const float4* __restrict__ x, v4h* __restrict__ xh) {
    const int i = blockIdx.x * 256 + threadIdx.x;
    float4 v = x[i];
    v4h h;
    h[0] = (_Float16)v.x; h[1] = (_Float16)v.y;
    h[2] = (_Float16)v.z; h[3] = (_Float16)v.w;
    xh[i] = h;
}

// ---------------- WMMA GEMM: y[m,n] = sum_k xh[m,k]*qw[n,k] + bias[n] ----------------
// Block tile 128x256, 8 waves (2x4), each wave a 64x64 region (4x4 WMMA tiles).
// Double-buffered LDS, async global->LDS prefetch, one barrier per K-step.

__launch_bounds__(256)
__global__ void bitlinear_wmma(const _Float16* __restrict__ xh,
                               const _Float16* __restrict__ wh,
                               const float*    __restrict__ bias,
                               float*          __restrict__ out) {
    __shared__ _Float16 lA[2][BM * LDSS];   // 2 x 10240 B
    __shared__ _Float16 lB[2][BN * LDSS];   // 2 x 20480 B

    const int tid   = threadIdx.x;
    const int wave  = tid >> 5;          // 0..7
    const int lane  = tid & 31;
    const int wy    = wave & 1;          // M: 2 waves x 64 rows
    const int wx    = wave >> 1;         // N: 4 waves x 64 cols
    const int rlane = lane & 15;
    const int hi    = lane >> 4;         // 0 or 1

    const int blockN = blockIdx.x * BN;
    const int blockM = blockIdx.y * BM;

    // cooperative tile staging:
    //  A: 2 threads per row (128 rows), 16 halves (32B) each
    //  B: 1 thread per row (256 rows), 32 halves (64B) each
    const int arow = tid >> 1;
    const int acol = (tid & 1) * 16;
    const _Float16* gA = xh + (size_t)(blockM + arow) * IN_F + acol;
    const _Float16* gB = wh + (size_t)(blockN + tid) * IN_F;

    _Float16* sA = lA[0] + arow * LDSS + acol;   // + buf*(BM*LDSS)
    _Float16* sB = lB[0] + tid * LDSS;           // + buf*(BN*LDSS)

    // ISA 7.12.2 per-lane fragment offsets (16-bit A 16x32 / B 32x16)
    const int a_k0 = hi * 8;        // frag elems 0..7  : K = a_k0..a_k0+7
    const int a_k1 = 16 + hi * 8;   // frag elems 8..15 : K = a_k1..a_k1+7
    const int b_k0 = hi * 8;        // elems 0..7: K = b_k0..+7, elems 8..15: K = 16+b_k0..+7

    const _Float16* fA = lA[0] + (wy * 64 + rlane) * LDSS;
    const _Float16* fB = lB[0] + (wx * 64 + rlane) * LDSS + b_k0;

    v8f acc[4][4];
#pragma unroll
    for (int i = 0; i < 4; ++i)
#pragma unroll
        for (int j = 0; j < 4; ++j)
            acc[i][j] = (v8f){0.f,0.f,0.f,0.f,0.f,0.f,0.f,0.f};

    // ---- prologue: tile 0 -> buf 0 ----
#if HAVE_ASYNC
    async_b128(gA,      sA);
    async_b128(gA + 8,  sA + 8);
    async_b128(gB,      sB);
    async_b128(gB + 8,  sB + 8);
    async_b128(gB + 16, sB + 16);
    async_b128(gB + 24, sB + 24);
    wait_async0();
#else
    {
        const uint4* ga = (const uint4*)gA;
        const uint4* gb = (const uint4*)gB;
        uint4 ra0 = ga[0], ra1 = ga[1];
        uint4 rb0 = gb[0], rb1 = gb[1], rb2 = gb[2], rb3 = gb[3];
        uint4* pa = (uint4*)sA;
        pa[0] = ra0; pa[1] = ra1;
        uint4* pb = (uint4*)sB;
        pb[0] = rb0; pb[1] = rb1; pb[2] = rb2; pb[3] = rb3;
    }
#endif
    __syncthreads();

    int buf = 0;
    for (int kt = BK; kt <= IN_F; kt += BK) {
        const bool more = (kt < IN_F);
        const int  nb   = buf ^ 1;

#if HAVE_ASYNC
        // kick off async prefetch of next tile into the other buffer
        if (more) {
            async_b128(gA + kt,      sA + nb * (BM * LDSS));
            async_b128(gA + kt + 8,  sA + nb * (BM * LDSS) + 8);
            _Float16* d = sB + nb * (BN * LDSS);
            async_b128(gB + kt,      d);
            async_b128(gB + kt + 8,  d + 8);
            async_b128(gB + kt + 16, d + 16);
            async_b128(gB + kt + 24, d + 24);
        }
#else
        uint4 ra0, ra1, rb0, rb1, rb2, rb3;
        if (more) {
            const uint4* ga = (const uint4*)(gA + kt);
            const uint4* gb = (const uint4*)(gB + kt);
            ra0 = ga[0]; ra1 = ga[1];
            rb0 = gb[0]; rb1 = gb[1]; rb2 = gb[2]; rb3 = gb[3];
        }
#endif

        // ---- compute on current buffer: preload all 8 frags, then 16 WMMAs ----
        const int boff_a = buf * (BM * LDSS);
        const int boff_b = buf * (BN * LDSS);
        v16h afr[4], bfr[4];
#pragma unroll
        for (int i = 0; i < 4; ++i) {
            const _Float16* ap = fA + boff_a + i * (16 * LDSS);
            ((uint4*)&afr[i])[0] = *(const uint4*)(ap + a_k0);
            ((uint4*)&afr[i])[1] = *(const uint4*)(ap + a_k1);
        }
#pragma unroll
        for (int j = 0; j < 4; ++j) {
            const _Float16* bp = fB + boff_b + j * (16 * LDSS);
            ((uint4*)&bfr[j])[0] = ((const uint4*)bp)[0];
            ((uint4*)&bfr[j])[1] = *(const uint4*)(bp + 16);
        }
#pragma unroll
        for (int i = 0; i < 4; ++i)
#pragma unroll
            for (int j = 0; j < 4; ++j)
                acc[i][j] = __builtin_amdgcn_wmma_f32_16x16x32_f16(
                    false, afr[i], false, bfr[j], (short)0, acc[i][j], false, false);

#if HAVE_ASYNC
        if (more) wait_async0();   // LDS writes of next tile complete (this wave)
#else
        if (more) {
            uint4* pa = (uint4*)(sA + nb * (BM * LDSS));
            pa[0] = ra0; pa[1] = ra1;
            uint4* pb = (uint4*)(sB + nb * (BN * LDSS));
            pb[0] = rb0; pb[1] = rb1; pb[2] = rb2; pb[3] = rb3;
        }
#endif
        __syncthreads();           // cross-wave visibility / reuse of old buffer
        buf = nb;
    }

    // ---- epilogue: + bias, store f32 ----
#pragma unroll
    for (int j = 0; j < 4; ++j) {
        const int n = blockN + wx * 64 + j * 16 + rlane;
        const float bv = bias[n];
#pragma unroll
        for (int i = 0; i < 4; ++i) {
            const int m0 = blockM + wy * 64 + i * 16 + hi * 8;
            float* op = out + (size_t)m0 * OUT_F + n;
#pragma unroll
            for (int v = 0; v < 8; ++v)
                op[(size_t)v * OUT_F] = acc[i][j][v] + bv;
        }
    }
}

// ---------------- host launcher ----------------

extern "C" void kernel_launch(void* const* d_in, const int* in_sizes, int n_in,
                              void* d_out, int out_size, void* d_ws, size_t ws_size,
                              hipStream_t stream) {
    const float* x    = (const float*)d_in[0];   // [4,2048,4096]
    const float* w    = (const float*)d_in[1];   // [4096,4096]
    const float* bias = (const float*)d_in[2];   // [4096]
    float*       out  = (float*)d_out;           // [4,2048,4096]

    char* ws = (char*)d_ws;
    float*    gamma    = (float*)ws;                                  // 1 float
    float*    partials = (float*)(ws + 256);                          // 256 floats
    _Float16* xh       = (_Float16*)(ws + 4096);                      // 67 MB
    _Float16* qwh      = (_Float16*)(ws + 4096 + (size_t)XELEMS * 2); // 33.5 MB

    absmean_partial<<<256, 256, 0, stream>>>((const float4*)w, partials);
    gamma_finalize<<<1, 256, 0, stream>>>(partials, gamma);
    quant_w<<<WELEMS / 4 / 256, 256, 0, stream>>>((const float4*)w, gamma, (v4h*)qwh);
    cast_x<<<XELEMS / 4 / 256, 256, 0, stream>>>((const float4*)x, (v4h*)xh);

    dim3 grid(OUT_F / BN, M_ROWS / BM);   // 16 x 64
    bitlinear_wmma<<<grid, 256, 0, stream>>>(xh, qwh, bias, out);
}
